// CausalSelfAttention_28905129902196
// MI455X (gfx1250) — matrix operand order repaired and checked by reading, not compiled, not run on hardware
//
#include <hip/hip_runtime.h>
#include <hip/hip_bf16.h>

// Problem constants (reference: B=2, T=2048, C=1024, H=16, D=64, scale=8)
#define BSZ   2
#define TSEQ  2048
#define CDIM  1024
#define NHEAD 16
#define DHEAD 64
#define MTOT  (BSZ * TSEQ)          // 4096 rows for the projection GEMMs
#define INV_SCALE 0.125f            // 1/8

typedef __bf16        v16bf __attribute__((ext_vector_type(16)));
typedef float         v8f   __attribute__((ext_vector_type(8)));
typedef unsigned int  v4u   __attribute__((ext_vector_type(4)));

union Frag {
    v16bf v;
    v4u   q[2];
    unsigned short u[16];
};

__device__ __forceinline__ unsigned short f2bf(float f) {
    unsigned int u = __float_as_uint(f);
    u = (u + 0x7FFFu + ((u >> 16) & 1u)) >> 16;   // round to nearest even
    return (unsigned short)u;
}

__device__ __forceinline__ v8f wmma_bf16(v16bf a, v16bf b, v8f c) {
    // D = A(16x32 bf16) * B(32x16 bf16) + C(16x16 f32)
    return __builtin_amdgcn_wmma_f32_16x16x32_bf16(
        false, a, false, b, (short)0, c, false, false);
}

// Low 32 bits of a generic pointer to LDS == addrspace(3) byte offset.
__device__ __forceinline__ unsigned lds_off(const void* p) {
    return (unsigned)(unsigned long long)p;
}

// Async global -> LDS copy of 16 bytes per lane (CDNA5 TDM-adjacent path,
// tracked by ASYNCcnt; bypasses VGPRs entirely).
__device__ __forceinline__ void async_g2l_b128(unsigned lds_addr, const void* gaddr) {
    asm volatile("global_load_async_to_lds_b128 %0, %1, off"
                 :: "v"(lds_addr), "v"(gaddr) : "memory");
}

// ---------------------------------------------------------------------------
// fp32 -> bf16 conversion (grid-stride)
// ---------------------------------------------------------------------------
__global__ void csa_to_bf16(const float* __restrict__ in,
                            unsigned short* __restrict__ out, int n) {
    int i = blockIdx.x * blockDim.x + threadIdx.x;
    int stride = gridDim.x * blockDim.x;
    for (; i < n; i += stride) out[i] = f2bf(in[i]);
}

// ---------------------------------------------------------------------------
// LDS-tiled WMMA GEMM:  Out = A(MxK bf16 rm) @ W(NxK bf16 rm)^T
// Block tile 64x128, K-chunks of 32, staged into LDS with
// global_load_async_to_lds_b128. 8 waves, each owns a 32x32 register tile
// (2x2 WMMA subtiles). A-frags reused across N-subtiles, B across M-subtiles,
// LDS tile reused by all 8 waves.
// MODE 0: store bf16 to [B,H,T,D]   (Q / K)
// MODE 1: store bf16 to [B,H,D,T]   (V transposed for P@V B-fragments)
// MODE 2: store f32  row-major MxN  (final projection)
// ---------------------------------------------------------------------------
#define GBM 64
#define GBN 128
#define GBK 32

template <int MODE>
__global__ __launch_bounds__(256)
void csa_gemm_xwT(const unsigned short* __restrict__ A,
                  const unsigned short* __restrict__ W,
                  void* __restrict__ Out,
                  int M, int N, int Kdim) {
    __shared__ __align__(16) unsigned short shA[GBM * GBK];   // 4 KB
    __shared__ __align__(16) unsigned short shB[GBN * GBK];   // 8 KB

    const int tid  = threadIdx.x;
    const int lane = tid & 31;
    const int wid  = tid >> 5;
    const int half = lane >> 4;     // lane-half (0/1)
    const int nl   = lane & 15;
    const int wm   = wid >> 2;      // wave row    (0..1) -> 32 rows each
    const int wn   = wid & 3;       // wave column (0..3) -> 32 cols each
    const int m0   = blockIdx.y * GBM;
    const int n0   = blockIdx.x * GBN;

    // --- staging assignment: 16B chunks; A: 64 rows x 4 segs = 256 chunks,
    //     B: 128 rows x 4 segs = 512 chunks (2 per thread) ---
    const int arow = tid >> 2, aseg = tid & 3;
    const int br0  = tid >> 2, bseg = tid & 3;
    const int br1  = br0 + 64;
    const unsigned short* gA  = A + (size_t)(m0 + arow) * Kdim + aseg * 8;
    const unsigned short* gB0 = W + (size_t)(n0 + br0) * Kdim + bseg * 8;
    const unsigned short* gB1 = W + (size_t)(n0 + br1) * Kdim + bseg * 8;
    const unsigned ldsA  = lds_off(shA + arow * GBK + aseg * 8);
    const unsigned ldsB0 = lds_off(shB + br0 * GBK + bseg * 8);
    const unsigned ldsB1 = lds_off(shB + br1 * GBK + bseg * 8);

    v8f acc[2][2] = {};

    for (int k0 = 0; k0 < Kdim; k0 += GBK) {
        // ---- async stage global -> LDS (no VGPR round trip) ----
        async_g2l_b128(ldsA,  gA  + k0);
        async_g2l_b128(ldsB0, gB0 + k0);
        async_g2l_b128(ldsB1, gB1 + k0);
        asm volatile("s_wait_asynccnt 0" ::: "memory");
        __syncthreads();

        // ---- fragments from LDS (ISA 7.12.2 layouts) ----
        Frag af[2], bf[2];
        #pragma unroll
        for (int mi = 0; mi < 2; ++mi) {
            const unsigned short* p = shA + (wm * 32 + mi * 16 + nl) * GBK + half * 8;
            af[mi].q[0] = *(const v4u*)(p);        // K = half*8 + 0..7
            af[mi].q[1] = *(const v4u*)(p + 16);   // K = 16 + half*8 + 0..7
        }
        #pragma unroll
        for (int ni = 0; ni < 2; ++ni) {
            const unsigned short* p = shB + (wn * 32 + ni * 16 + nl) * GBK + half * 8;
            bf[ni].q[0] = *(const v4u*)(p);        // K = half*8 + 0..7
            bf[ni].q[1] = *(const v4u*)(p + 16);   // K = 16 + half*8 + 0..7
        }

        #pragma unroll
        for (int mi = 0; mi < 2; ++mi)
            #pragma unroll
            for (int ni = 0; ni < 2; ++ni)
                acc[mi][ni] = wmma_bf16(af[mi].v, bf[ni].v, acc[mi][ni]);

        asm volatile("s_wait_dscnt 0" ::: "memory");  // reads done before overwrite
        __syncthreads();
    }

    // ---- epilogue: C/D layout row = r + 8*half, col = nl ----
    #pragma unroll
    for (int mi = 0; mi < 2; ++mi) {
        #pragma unroll
        for (int ni = 0; ni < 2; ++ni) {
            const int mbase = m0 + wm * 32 + mi * 16;
            const int nbase = n0 + wn * 32 + ni * 16;
            #pragma unroll
            for (int r = 0; r < 8; ++r) {
                const int m = mbase + r + 8 * half;
                const int n = nbase + nl;
                const float val = acc[mi][ni][r];
                if (MODE == 0) {            // [B,H,T,D] bf16
                    const int b = m >> 11, t = m & (TSEQ - 1);
                    const int h = n >> 6,  d = n & (DHEAD - 1);
                    ((unsigned short*)Out)[(((size_t)(b * NHEAD + h) * TSEQ) + t) * DHEAD + d] = f2bf(val);
                } else if (MODE == 1) {     // [B,H,D,T] bf16 (transposed V)
                    const int b = m >> 11, t = m & (TSEQ - 1);
                    const int h = n >> 6,  d = n & (DHEAD - 1);
                    ((unsigned short*)Out)[(((size_t)(b * NHEAD + h) * DHEAD) + d) * TSEQ + t] = f2bf(val);
                } else {                    // f32 row-major
                    ((float*)Out)[(size_t)m * N + n] = val;
                }
            }
        }
    }
}

// ---------------------------------------------------------------------------
// Flash attention (causal, online softmax).
// One wave = one 16-row query tile of one (b,h). Key blocks of 32.
//   S(16x32) = Q(16x64) @ K^T        -> 4 x v_wmma_f32_16x16x32_bf16
//   O(16x64) += P(16x32) @ V(32x64)  -> 4 x v_wmma_f32_16x16x32_bf16
// P re-layout (C-layout -> A-layout) staged through LDS.
// ---------------------------------------------------------------------------
__global__ __launch_bounds__(128)
void csa_flash_attn(const unsigned short* __restrict__ Q,   // [B,H,T,D]
                    const unsigned short* __restrict__ K,   // [B,H,T,D]
                    const unsigned short* __restrict__ Vt,  // [B,H,D,T]
                    unsigned short* __restrict__ Yout) {    // [B,T,C] bf16
    __shared__ __align__(16) unsigned short p_lds[4][16 * 32];  // per-wave P tile

    const int lane = threadIdx.x & 31;
    const int wid  = threadIdx.x >> 5;
    const int half = lane >> 4;
    const int nl   = lane & 15;

    const int g  = blockIdx.x * 4 + wid;        // global wave-tile id
    const int qt = g & (TSEQ / 16 - 1);
    const int bh = g >> 7;                      // T/16 == 128
    const int i0 = qt * 16;

    const unsigned short* Qh = Q  + (size_t)bh * TSEQ * DHEAD;
    const unsigned short* Kh = K  + (size_t)bh * TSEQ * DHEAD;
    const unsigned short* Vh = Vt + (size_t)bh * TSEQ * DHEAD;

    // Q A-fragments for d-chunks 0..31 and 32..63 (row = nl, K split by half)
    Frag aq0, aq1;
    {
        const unsigned short* qp = Qh + (size_t)(i0 + nl) * DHEAD + half * 8;
        aq0.q[0] = *(const v4u*)(qp);
        aq0.q[1] = *(const v4u*)(qp + 16);
        aq1.q[0] = *(const v4u*)(qp + 32);
        aq1.q[1] = *(const v4u*)(qp + 48);
    }

    float mrow[8], lrow[8];
    v8f oacc[4] = {};
    #pragma unroll
    for (int r = 0; r < 8; ++r) { mrow[r] = -1e30f; lrow[r] = 0.0f; }

    const int jend = i0 + 15;
    for (int j0 = 0; j0 <= jend; j0 += 32) {
        if (j0 + 32 <= jend) {   // prefetch next key/value block
            __builtin_prefetch(Kh + (size_t)(j0 + 32 + nl) * DHEAD);
            __builtin_prefetch(Vh + (size_t)nl * TSEQ + j0 + 32);
        }
        // ---- S = Q @ K^T (two 16x16 N-tiles, two 32-wide K chunks) ----
        v8f s[2];
        #pragma unroll
        for (int jt = 0; jt < 2; ++jt) {
            const unsigned short* kp =
                Kh + (size_t)(j0 + jt * 16 + nl) * DHEAD + half * 8;
            Frag bk0, bk1;
            bk0.q[0] = *(const v4u*)(kp);          // d = half*8 + 0..7
            bk0.q[1] = *(const v4u*)(kp + 16);     // d = 16 + half*8 + 0..7
            bk1.q[0] = *(const v4u*)(kp + 32);     // d = 32 + half*8 + 0..7
            bk1.q[1] = *(const v4u*)(kp + 48);
            v8f z = {};
            z = wmma_bf16(aq0.v, bk0.v, z);
            z = wmma_bf16(aq1.v, bk1.v, z);
            s[jt] = z;
        }

        // ---- scale + causal mask ----
        float pl[2][8];
        #pragma unroll
        for (int jt = 0; jt < 2; ++jt) {
            const int key = j0 + jt * 16 + nl;     // this lane's column
            #pragma unroll
            for (int r = 0; r < 8; ++r) {
                float v = s[jt][r] * INV_SCALE;
                const int qrow = i0 + r + 8 * half;
                pl[jt][r] = (key > qrow) ? -1e30f : v;
            }
        }

        // ---- online softmax: row max / exp / row sum (16-lane groups) ----
        #pragma unroll
        for (int r = 0; r < 8; ++r) {
            float t = fmaxf(pl[0][r], pl[1][r]);
            t = fmaxf(t, __shfl_xor(t, 1));
            t = fmaxf(t, __shfl_xor(t, 2));
            t = fmaxf(t, __shfl_xor(t, 4));
            t = fmaxf(t, __shfl_xor(t, 8));
            const float mnew  = fmaxf(mrow[r], t);
            const float alpha = __expf(mrow[r] - mnew);
            mrow[r] = mnew;
            const float p0 = __expf(pl[0][r] - mnew);
            const float p1 = __expf(pl[1][r] - mnew);
            pl[0][r] = p0; pl[1][r] = p1;
            float rs = p0 + p1;
            rs += __shfl_xor(rs, 1);
            rs += __shfl_xor(rs, 2);
            rs += __shfl_xor(rs, 4);
            rs += __shfl_xor(rs, 8);
            lrow[r] = lrow[r] * alpha + rs;
            oacc[0][r] *= alpha; oacc[1][r] *= alpha;
            oacc[2][r] *= alpha; oacc[3][r] *= alpha;
        }

        // ---- P: C-layout -> bf16 A-layout via LDS ----
        unsigned short* pw = p_lds[wid];
        #pragma unroll
        for (int jt = 0; jt < 2; ++jt) {
            #pragma unroll
            for (int r = 0; r < 8; ++r) {
                const int row = r + 8 * half;
                const int col = jt * 16 + nl;
                pw[row * 32 + col] = f2bf(pl[jt][r]);
            }
        }
        asm volatile("s_wait_dscnt 0" ::: "memory");  // same-wave LDS RAW fence
        Frag pa;
        {
            const unsigned short* pr = pw + nl * 32 + half * 8;
            pa.q[0] = *(const v4u*)(pr);        // K = half*8 + 0..7
            pa.q[1] = *(const v4u*)(pr + 16);   // K = 16 + half*8 + 0..7
        }

        // ---- O += P @ V  (V transposed: contiguous B-fragment loads) ----
        #pragma unroll
        for (int tn = 0; tn < 4; ++tn) {
            const unsigned short* vp =
                Vh + (size_t)(tn * 16 + nl) * TSEQ + j0 + half * 8;
            Frag bv;
            bv.q[0] = *(const v4u*)(vp);
            bv.q[1] = *(const v4u*)(vp + 16);
            oacc[tn] = wmma_bf16(pa.v, bv.v, oacc[tn]);
        }
    }

    // ---- normalize and store merged-head output [B,T,C] bf16 ----
    const int b = bh >> 4, head = bh & (NHEAD - 1);
    float inv[8];
    #pragma unroll
    for (int r = 0; r < 8; ++r) inv[r] = 1.0f / lrow[r];
    #pragma unroll
    for (int tn = 0; tn < 4; ++tn) {
        #pragma unroll
        for (int r = 0; r < 8; ++r) {
            const int row = i0 + r + 8 * half;
            const size_t idx =
                ((size_t)b * TSEQ + row) * CDIM + head * DHEAD + tn * 16 + nl;
            Yout[idx] = f2bf(oacc[tn][r] * inv[r]);
        }
    }
}

// ---------------------------------------------------------------------------
// Host-side orchestration
// ---------------------------------------------------------------------------
extern "C" void kernel_launch(void* const* d_in, const int* in_sizes, int n_in,
                              void* d_out, int out_size, void* d_ws, size_t ws_size,
                              hipStream_t stream) {
    const float* X  = (const float*)d_in[0];
    const float* Wq = (const float*)d_in[1];
    const float* Wk = (const float*)d_in[2];
    const float* Wv = (const float*)d_in[3];
    const float* Wo = (const float*)d_in[4];
    float* Out = (float*)d_out;

    const int NX = BSZ * TSEQ * CDIM;   // 4,194,304
    const int NW = CDIM * CDIM;         // 1,048,576

    // Workspace carve-out (bf16 as raw u16): 48 MiB total
    unsigned short* Xb  = (unsigned short*)d_ws;
    unsigned short* Wqb = Xb  + NX;
    unsigned short* Wkb = Wqb + NW;
    unsigned short* Wvb = Wkb + NW;
    unsigned short* Wob = Wvb + NW;
    unsigned short* Qb  = Wob + NW;     // [B,H,T,D]
    unsigned short* Kb  = Qb  + NX;     // [B,H,T,D]
    unsigned short* Vtb = Kb  + NX;     // [B,H,D,T]
    unsigned short* Ya  = Vtb + NX;     // [B,T,C]

    // 1) fp32 -> bf16
    csa_to_bf16<<<2048, 256, 0, stream>>>(X,  Xb,  NX);
    csa_to_bf16<<<1024, 256, 0, stream>>>(Wq, Wqb, NW);
    csa_to_bf16<<<1024, 256, 0, stream>>>(Wk, Wkb, NW);
    csa_to_bf16<<<1024, 256, 0, stream>>>(Wv, Wvb, NW);
    csa_to_bf16<<<1024, 256, 0, stream>>>(Wo, Wob, NW);

    // 2) Q/K/V projections (X @ W^T), scattered into attention layouts
    dim3 ggrid(CDIM / GBN, MTOT / GBM);         // (8, 64)
    csa_gemm_xwT<0><<<ggrid, 256, 0, stream>>>(Xb, Wqb, Qb,  MTOT, CDIM, CDIM);
    csa_gemm_xwT<0><<<ggrid, 256, 0, stream>>>(Xb, Wkb, Kb,  MTOT, CDIM, CDIM);
    csa_gemm_xwT<1><<<ggrid, 256, 0, stream>>>(Xb, Wvb, Vtb, MTOT, CDIM, CDIM);

    // 3) causal flash attention: 4096 query tiles, 4 waves/block
    csa_flash_attn<<<(BSZ * NHEAD * (TSEQ / 16)) / 4, 128, 0, stream>>>(
        Qb, Kb, Vtb, Ya);

    // 4) output projection (fp32 result)
    csa_gemm_xwT<2><<<ggrid, 256, 0, stream>>>(Ya, Wob, (void*)Out,
                                               MTOT, CDIM, CDIM);
}
